// GemmaAttention_77610059039032
// MI455X (gfx1250) — compile-verified, matrix operand order checked by reading, not run on hardware
//
#include <hip/hip_runtime.h>
#include <hip/hip_bf16.h>

// ---------------------------------------------------------------------------
// GemmaAttention forward for MI455X (gfx1250), wave32 + WMMA bf16 path.
//
//   weight pre-swizzle (one-time, fp32 -> bf16 fragment order)
//   -> QKV gemms (128x128 tile, v_wmma_f32_16x16x32_bf16, async LDS staging)
//   -> RoPE (writes bf16, fragment-swizzled)
//   -> flash attention (async K/V staging, WMMA, online softmax)
//   -> output projection gemm
//
// All tiles live in LDS pre-swizzled into the WMMA fragment layout so every
// fragment is a contiguous 32B span (2x ds_load_b128), and all global->LDS
// staging is 16B-granular GLOBAL_LOAD_ASYNC_TO_LDS_B128 tracked by
// ASYNCcnt / s_wait_asynccnt.
// ---------------------------------------------------------------------------

#define NUM_HEADS    16
#define NUM_KV_HEADS 4
#define HEAD_DIM     256
#define SEQ          2048           // fixed problem size (S)

typedef __attribute__((ext_vector_type(16))) __bf16 v16bf;
typedef __attribute__((ext_vector_type(8)))  __bf16 v8bf;
typedef __attribute__((ext_vector_type(8)))  float  v8f;
typedef int v4i __attribute__((vector_size(16)));

#if defined(__HIP_DEVICE_COMPILE__) && __has_builtin(__builtin_amdgcn_global_load_async_to_lds_b128)
#define HAVE_ASYNC_LDS 1
typedef __attribute__((address_space(1))) v4i* g_v4i_ptr;
typedef __attribute__((address_space(3))) v4i* l_v4i_ptr;
#else
#define HAVE_ASYNC_LDS 0
#endif

// Fragment K-mapping (ISA 7.12.2, 16-bit A/B): lane half h, element e ->
//   k = e + 8h (e<8) ; k = e + 8 + 8h (e>=8).
// As an in-row permutation this is a bit3<->bit4 swap: an involution.
__device__ __forceinline__ int swz32(int kk) {
    return ((kk & 8) << 1) | ((kk & 16) >> 1) | (kk & 7);
}
// swz32 on a full column index, permuting within its 32-aligned group.
__device__ __forceinline__ int swzcol(int c) {
    return (c & ~31) | swz32(c & 31);
}
// Aligned 8-col group g -> aligned 8-elem dest base under swz32.
__device__ __forceinline__ int grp_base(int g) {
    return ((g & 1) << 4) | ((g & 2) << 2);
}

__device__ __forceinline__ v8bf pack8(float4 f0, float4 f1) {
    v8bf p;
    p[0] = (__bf16)f0.x; p[1] = (__bf16)f0.y; p[2] = (__bf16)f0.z; p[3] = (__bf16)f0.w;
    p[4] = (__bf16)f1.x; p[5] = (__bf16)f1.y; p[6] = (__bf16)f1.z; p[7] = (__bf16)f1.w;
    return p;
}

// 16B global->LDS copy: async (ASYNCcnt) when the builtin exists.
__device__ __forceinline__ void cp16_g2l(const __bf16* g, __bf16* l) {
#if HAVE_ASYNC_LDS
    __builtin_amdgcn_global_load_async_to_lds_b128(
        (g_v4i_ptr)(uintptr_t)(const void*)g,
        (l_v4i_ptr)(uintptr_t)(const void*)l,
        0, 0);
#else
    *(v8bf*)l = *(const v8bf*)g;
#endif
}
__device__ __forceinline__ void async_join() {
#if HAVE_ASYNC_LDS
    asm volatile("s_wait_asynccnt 0" ::: "memory");
#endif
}

// ---------------------------------------------------------------------------
// One-time weight pre-swizzle: W[K][N] fp32 -> Wz bf16 in fragment order:
//   Wz[((k>>5)*N + n)*32 + swz32(k&31)] = W[k][n]
// A (k0,n0) tile column is then 32 contiguous bf16 (64B) per output column.
// ---------------------------------------------------------------------------
__global__ void weight_swz_kernel(const float* __restrict__ W,
                                  __bf16* __restrict__ Wz, int K, int N)
{
    int idx = blockIdx.x * 256 + threadIdx.x;
    if (idx >= K * N) return;
    int n = idx % N;
    int k = idx / N;
    Wz[((size_t)(k >> 5) * N + n) * 32 + swz32(k & 31)] = (__bf16)W[idx];
}

// ---------------------------------------------------------------------------
// GEMM: C[M,N] = A[M,K] @ W[K,N]; A fp32-plain or bf16-preswizzled, W is Wz.
// Block tile 128x128, K-step 32, 256 threads = 8 waves (4x2); each wave owns
// a 32x64 sub-tile -> 8 v_wmma_f32_16x16x32_bf16 per K-step.
// OMODE 0: fp32 plain C.  OMODE 1: bf16 V-transposed+swizzled C (for V).
// ---------------------------------------------------------------------------
#define GBM 128
#define GBN 128
#define GBK 32
#define FR_STR 48   // padded fragment-row stride (elements): 96B, 32B-aligned

template <typename AT, int OMODE>
__global__ __launch_bounds__(256) void gemm_bf16_wmma_kernel(
    const AT* __restrict__ A, const __bf16* __restrict__ Wz,
    void* Cv, int M, int N, int K)
{
    __shared__ __bf16 As[GBM * FR_STR];   // 12 KB
    __shared__ __bf16 Ws[GBN * FR_STR];   // 12 KB

    const int tid  = threadIdx.x;
    const int lane = tid & 31;
    const int wave = tid >> 5;
    const int wm   = wave & 3;     // 32-row slice
    const int wn   = wave >> 2;    // 64-col slice
    const int h    = lane >> 4;
    const int ml   = lane & 15;

    const int m0 = blockIdx.y * GBM;
    const int n0 = blockIdx.x * GBN;

    v8f acc[2][4];
    #pragma unroll
    for (int i = 0; i < 2; ++i)
        #pragma unroll
        for (int j = 0; j < 4; ++j) acc[i][j] = (v8f){};

    for (int k0 = 0; k0 < K; k0 += GBK) {
        // ---- Stage A tile (128x32).
        if constexpr (sizeof(AT) == 4) {
            for (int c8 = tid; c8 < (GBM * GBK) / 8; c8 += 256) {
                int r = c8 >> 2, g = c8 & 3;
                const float* src = (const float*)&A[(size_t)(m0 + r) * K + k0 + g * 8];
                float4 f0 = *(const float4*)src;
                float4 f1 = *(const float4*)(src + 4);
                *(v8bf*)&As[r * FR_STR + grp_base(g)] = pack8(f0, f1);
            }
        } else {
            // bf16 pre-swizzled rows: pure 16B copies.
            for (int c16 = tid; c16 < GBM * 4; c16 += 256) {
                int r = c16 >> 2, co = (c16 & 3) * 8;
                cp16_g2l((const __bf16*)&A[(size_t)(m0 + r) * K + k0 + co],
                         &As[r * FR_STR + co]);
            }
        }
        // ---- Stage W tile (32x128) from pre-swizzled Wz: pure 16B copies.
        for (int c16 = tid; c16 < GBN * 4; c16 += 256) {
            int c = c16 >> 2, co = (c16 & 3) * 8;
            cp16_g2l(&Wz[((size_t)(k0 >> 5) * N + n0 + c) * 32 + co],
                     &Ws[c * FR_STR + co]);
        }
        async_join();
        __syncthreads();

        v16bf af0 = *(const v16bf*)&As[(wm * 32 + ml) * FR_STR + h * 16];
        v16bf af1 = *(const v16bf*)&As[(wm * 32 + 16 + ml) * FR_STR + h * 16];
        #pragma unroll
        for (int fc = 0; fc < 4; ++fc) {
            v16bf bfr = *(const v16bf*)&Ws[(wn * 64 + fc * 16 + ml) * FR_STR + h * 16];
            acc[0][fc] = __builtin_amdgcn_wmma_f32_16x16x32_bf16(
                false, af0, false, bfr, (short)0, acc[0][fc], false, false);
            acc[1][fc] = __builtin_amdgcn_wmma_f32_16x16x32_bf16(
                false, af1, false, bfr, (short)0, acc[1][fc], false, false);
        }
        __syncthreads();
    }

    // ---- Epilogue.
    #pragma unroll
    for (int fr = 0; fr < 2; ++fr) {
        #pragma unroll
        for (int fc = 0; fc < 4; ++fc) {
            #pragma unroll
            for (int r = 0; r < 8; ++r) {
                int row = m0 + wm * 32 + fr * 16 + h * 8 + r;
                int col = n0 + wn * 64 + fc * 16 + ml;
                if constexpr (OMODE == 0) {
                    ((float*)Cv)[(size_t)row * N + col] = acc[fr][fc][r];
                } else {
                    // V output: [b][kvh][d][s] with s swizzled in 32-groups.
                    int bb = row >> 11;          // row / SEQ
                    int ss = row & (SEQ - 1);
                    size_t idx = ((size_t)bb * N + col) * SEQ + swzcol(ss);
                    ((__bf16*)Cv)[idx] = (__bf16)acc[fr][fc][r];
                }
            }
        }
    }
}

// ---------------------------------------------------------------------------
// RoPE producers: fp32 in, bf16 fragment-swizzled out.
// ---------------------------------------------------------------------------
__global__ void rope_q_kernel(const float* __restrict__ Qf,
                              __bf16* __restrict__ Qbf,
                              const int* __restrict__ position_ids, int rows)
{
    int idx = blockIdx.x * 256 + threadIdx.x;
    if (idx >= rows * NUM_HEADS * 128) return;
    int d    = idx & 127;
    int t    = idx >> 7;
    int head = t % NUM_HEADS;
    int row  = t / NUM_HEADS;

    float pos = (float)position_ids[row];
    float inv_freq = __powf(10000.0f, -((float)(2 * d)) / 256.0f);
    float s, c;
    __sincosf(pos * inv_freq, &s, &c);

    const float* p = Qf + (size_t)row * (NUM_HEADS * HEAD_DIM) + head * HEAD_DIM;
    __bf16* q = Qbf + (size_t)row * (NUM_HEADS * HEAD_DIM) + head * HEAD_DIM;
    float x1 = p[d], x2 = p[d + 128];
    q[swzcol(d)]       = (__bf16)(x1 * c - x2 * s);
    q[swzcol(d + 128)] = (__bf16)(x2 * c + x1 * s);
}

__global__ void rope_k_kernel(const float* __restrict__ Kf,
                              __bf16* __restrict__ Kswz,   // [b][kvh][s][256 swz]
                              const int* __restrict__ position_ids, int rows)
{
    int idx = blockIdx.x * 256 + threadIdx.x;
    if (idx >= rows * NUM_KV_HEADS * 128) return;
    int d   = idx & 127;
    int t   = idx >> 7;
    int kvh = t % NUM_KV_HEADS;
    int row = t / NUM_KV_HEADS;
    int b   = row >> 11;            // row / SEQ
    int s_  = row & (SEQ - 1);

    float pos = (float)position_ids[row];
    float inv_freq = __powf(10000.0f, -((float)(2 * d)) / 256.0f);
    float sn, cs;
    __sincosf(pos * inv_freq, &sn, &cs);

    const float* p = Kf + (size_t)row * (NUM_KV_HEADS * HEAD_DIM) + kvh * HEAD_DIM;
    __bf16* ko = Kswz + (((size_t)b * NUM_KV_HEADS + kvh) * SEQ + s_) * HEAD_DIM;
    float x1 = p[d], x2 = p[d + 128];
    ko[swzcol(d)]       = (__bf16)(x1 * cs - x2 * sn);
    ko[swzcol(d + 128)] = (__bf16)(x2 * cs + x1 * sn);
}

// ---------------------------------------------------------------------------
// Flash attention (causal, GQA). Grid: (B*NUM_HEADS, S/128), 256 threads.
// Q: bf16 swizzled [token][4096]; K: bf16 swizzled [b][kvh][s][256];
// V: bf16 transposed [b][kvh][d][s-swz]. Output: bf16 swizzled [token][4096].
// ---------------------------------------------------------------------------
#define QTILE 128
#define KVT   32
#define KS_STR 272   // per-kv-row stride for K tile (544B, 32B-aligned)

__global__ __launch_bounds__(256) void flash_attn_wmma_kernel(
    const __bf16* __restrict__ Q,
    const __bf16* __restrict__ Kswz,
    const __bf16* __restrict__ Vt,
    __bf16* __restrict__ O,
    int B, int S)
{
    const int bh     = blockIdx.x;
    const int b      = bh / NUM_HEADS;
    const int head   = bh % NUM_HEADS;
    const int kvhead = head / (NUM_HEADS / NUM_KV_HEADS);
    const int qt     = blockIdx.y;

    __shared__ __bf16 Ks[KVT * KS_STR];        // K tile, fragment order (~17 KB)
    __shared__ __bf16 Vs[HEAD_DIM * FR_STR];   // V tile [d][swz kv]   (~24 KB)
    __shared__ __bf16 Ps[8 * 16 * FR_STR];     // per-wave P relayout  (~12 KB)

    const int tid  = threadIdx.x;
    const int lane = tid & 31;
    const int wave = tid >> 5;
    const int h    = lane >> 4;
    const int ml   = lane & 15;

    const int q0 = qt * QTILE + wave * 16;
    const size_t qstride = NUM_HEADS * HEAD_DIM;
    const __bf16* qrow = Q + ((size_t)b * S + q0) * qstride + head * HEAD_DIM
                           + (size_t)ml * qstride;

    // Q fragments: one aligned 32B read each (data pre-swizzled in global).
    v16bf qf[8];
    #pragma unroll
    for (int ks = 0; ks < 8; ++ks)
        qf[ks] = *(const v16bf*)(qrow + ks * 32 + h * 16);

    v8f o[16];
    #pragma unroll
    for (int i = 0; i < 16; ++i) o[i] = (v8f){};
    float mrow[8], lrow[8];
    #pragma unroll
    for (int r = 0; r < 8; ++r) { mrow[r] = -1e30f; lrow[r] = 0.0f; }

    const float scale = 0.0625f;               // 1/sqrt(256)
    const int kv_end_block = qt * QTILE + QTILE;
    const int kv_end_wave  = q0 + 16;

    const __bf16* kbase = Kswz + ((size_t)b * NUM_KV_HEADS + kvhead) * (size_t)S * HEAD_DIM;
    const __bf16* vbase = Vt   + ((size_t)b * NUM_KV_HEADS + kvhead) * (size_t)HEAD_DIM * S;

    const int sml0 = swz32(ml);
    const int sml1 = swz32(16 + ml);
    const int pbase = wave * 16 * FR_STR;

    for (int kv0 = 0; kv0 < kv_end_block; kv0 += KVT) {
        __syncthreads();
        // ---- Async-stage K tile: 32 rows x 512B = 1024 16B chunks.
        for (int c16 = tid; c16 < 1024; c16 += 256) {
            int r  = c16 >> 5;
            int co = (c16 & 31) * 8;
            cp16_g2l(kbase + (size_t)(kv0 + r) * HEAD_DIM + co,
                     &Ks[r * KS_STR + co]);
        }
        // ---- Async-stage V tile: 256 d-rows x 64B = 1024 16B chunks.
        for (int c16 = tid; c16 < 1024; c16 += 256) {
            int d  = c16 >> 2;
            int co = (c16 & 3) * 8;
            cp16_g2l(vbase + (size_t)d * S + kv0 + co,
                     &Vs[d * FR_STR + co]);
        }
        if (kv0 + KVT < kv_end_block)
            __builtin_prefetch(kbase + (size_t)(kv0 + KVT + (tid >> 3)) * HEAD_DIM, 0, 0);
        async_join();
        __syncthreads();

        if (kv0 >= kv_end_wave) continue;

        // ---- S = Q*K^T over d=256: two 16x16 score accumulators.
        v8f s0 = {}, s1 = {};
        #pragma unroll
        for (int ks = 0; ks < 8; ++ks) {
            v16bf b0 = *(const v16bf*)&Ks[ml * KS_STR + ks * 32 + h * 16];
            v16bf b1 = *(const v16bf*)&Ks[(16 + ml) * KS_STR + ks * 32 + h * 16];
            s0 = __builtin_amdgcn_wmma_f32_16x16x32_bf16(
                false, qf[ks], false, b0, (short)0, s0, false, false);
            s1 = __builtin_amdgcn_wmma_f32_16x16x32_bf16(
                false, qf[ks], false, b1, (short)0, s1, false, false);
        }

        // ---- Online softmax (rows r+8h, cols ml / 16+ml).
        float p0[8], p1[8];
        #pragma unroll
        for (int r = 0; r < 8; ++r) {
            int qrowi = q0 + r + 8 * h;
            float v0 = s0[r] * scale;
            float v1 = s1[r] * scale;
            if (kv0 + ml > qrowi)      v0 = -1e30f;
            if (kv0 + 16 + ml > qrowi) v1 = -1e30f;
            float mx = fmaxf(v0, v1);
            #pragma unroll
            for (int off = 1; off < 16; off <<= 1)
                mx = fmaxf(mx, __shfl_xor(mx, off, 32));
            float mnew  = fmaxf(mrow[r], mx);
            float alpha = __expf(mrow[r] - mnew);
            float e0 = __expf(v0 - mnew);
            float e1 = __expf(v1 - mnew);
            float rs = e0 + e1;
            #pragma unroll
            for (int off = 1; off < 16; off <<= 1)
                rs += __shfl_xor(rs, off, 32);
            lrow[r] = lrow[r] * alpha + rs;
            mrow[r] = mnew;
            p0[r] = e0;
            p1[r] = e1;
            #pragma unroll
            for (int dc = 0; dc < 16; ++dc) o[dc][r] *= alpha;
        }

        // ---- Relayout P (C-layout -> A-fragment) through per-wave LDS.
        #pragma unroll
        for (int r = 0; r < 8; ++r) {
            Ps[pbase + (r + 8 * h) * FR_STR + sml0] = (__bf16)p0[r];
            Ps[pbase + (r + 8 * h) * FR_STR + sml1] = (__bf16)p1[r];
        }
        asm volatile("s_wait_dscnt 0" ::: "memory");
        v16bf pf = *(const v16bf*)&Ps[pbase + ml * FR_STR + h * 16];

        // ---- O(16x256) += P(16x32) * V(32x256).
        #pragma unroll
        for (int dc = 0; dc < 16; ++dc) {
            v16bf vf = *(const v16bf*)&Vs[(dc * 16 + ml) * FR_STR + h * 16];
            o[dc] = __builtin_amdgcn_wmma_f32_16x16x32_bf16(
                false, pf, false, vf, (short)0, o[dc], false, false);
        }
    }

    // ---- Normalize, store bf16 with fragment swizzle (for out-proj GEMM).
    __bf16* optr = O + ((size_t)b * S + q0) * qstride + head * HEAD_DIM;
    #pragma unroll
    for (int r = 0; r < 8; ++r) {
        float inv = 1.0f / lrow[r];
        int row = r + 8 * h;
        #pragma unroll
        for (int dc = 0; dc < 16; ++dc) {
            int off = (dc >> 1) * 32 + ((dc & 1) ? sml1 : sml0);
            optr[(size_t)row * qstride + off] = (__bf16)(o[dc][r] * inv);
        }
    }
}

// ---------------------------------------------------------------------------
// Host-side launch
// ---------------------------------------------------------------------------
extern "C" void kernel_launch(void* const* d_in, const int* in_sizes, int n_in,
                              void* d_out, int out_size, void* d_ws, size_t ws_size,
                              hipStream_t stream)
{
    (void)in_sizes; (void)n_in; (void)out_size; (void)ws_size;

    const float* hidden  = (const float*)d_in[0];
    const int*   pos_ids = (const int*)  d_in[1];
    const float* wq      = (const float*)d_in[2];
    const float* wk      = (const float*)d_in[3];
    const float* wv      = (const float*)d_in[4];
    const float* wo      = (const float*)d_in[5];
    float*       out     = (float*)d_out;

    const int B = 2, S = SEQ, H = 2048;
    const int M  = B * S;                        // 4096 token rows
    const int NQ = NUM_HEADS * HEAD_DIM;         // 4096
    const int NK = NUM_KV_HEADS * HEAD_DIM;      // 1024

    // Workspace layout (~176 MB).
    char* ws = (char*)d_ws;
    float*  Qb   = (float*)ws;  ws += (size_t)M * NQ * 4;    // fp32 Q (pre-RoPE)
    float*  Kb   = (float*)ws;  ws += (size_t)M * NK * 4;    // fp32 K (pre-RoPE)
    __bf16* Qbf  = (__bf16*)ws; ws += (size_t)M * NQ * 2;    // bf16 swz Q
    __bf16* Kswz = (__bf16*)ws; ws += (size_t)M * NK * 2;    // bf16 swz K
    __bf16* Vt   = (__bf16*)ws; ws += (size_t)M * NK * 2;    // bf16 V transposed
    __bf16* Wzq  = (__bf16*)ws; ws += (size_t)H * NQ * 2;    // pre-swizzled weights
    __bf16* Wzk  = (__bf16*)ws; ws += (size_t)H * NK * 2;
    __bf16* Wzv  = (__bf16*)ws; ws += (size_t)H * NK * 2;
    __bf16* Wzo  = (__bf16*)ws;
    __bf16* Ab   = (__bf16*)Qb;   // alias: fp32 Q is dead after RoPE

    dim3 blk(256);

    // One-time weight pre-swizzle (fp32 -> bf16 fragment order).
    weight_swz_kernel<<<(H * NQ + 255) / 256, blk, 0, stream>>>(wq, Wzq, H, NQ);
    weight_swz_kernel<<<(H * NK + 255) / 256, blk, 0, stream>>>(wk, Wzk, H, NK);
    weight_swz_kernel<<<(H * NK + 255) / 256, blk, 0, stream>>>(wv, Wzv, H, NK);
    weight_swz_kernel<<<(NQ * H + 255) / 256, blk, 0, stream>>>(wo, Wzo, NQ, H);

    // QKV projections (V written transposed+swizzled bf16).
    gemm_bf16_wmma_kernel<float, 0><<<dim3(NQ / GBN, M / GBM), blk, 0, stream>>>(
        hidden, Wzq, (void*)Qb, M, NQ, H);
    gemm_bf16_wmma_kernel<float, 0><<<dim3(NK / GBN, M / GBM), blk, 0, stream>>>(
        hidden, Wzk, (void*)Kb, M, NK, H);
    gemm_bf16_wmma_kernel<float, 1><<<dim3(NK / GBN, M / GBM), blk, 0, stream>>>(
        hidden, Wzv, (void*)Vt, M, NK, H);

    // RoPE -> bf16 swizzled Q/K.
    rope_q_kernel<<<(M * NUM_HEADS * 128 + 255) / 256, blk, 0, stream>>>(
        Qb, Qbf, pos_ids, M);
    rope_k_kernel<<<(M * NUM_KV_HEADS * 128 + 255) / 256, blk, 0, stream>>>(
        Kb, Kswz, pos_ids, M);

    // Causal GQA flash attention.
    flash_attn_wmma_kernel<<<dim3(B * NUM_HEADS, S / QTILE), blk, 0, stream>>>(
        Qbf, Kswz, Vt, Ab, B, S);

    // Output projection (bf16 A, pre-swizzled).
    gemm_bf16_wmma_kernel<__bf16, 0><<<dim3(H / GBN, M / GBM), blk, 0, stream>>>(
        Ab, Wzo, (void*)out, M, H, NQ);
}
